// CausalSelfAttention_36730560315818
// MI455X (gfx1250) — compile-verified, matrix operand order checked by reading, not run on hardware
//
#include <hip/hip_runtime.h>
#include <hip/hip_bf16.h>

#define DIMC  2048
#define NHEAD 16
#define HDIM  128
#define BATCH 2
#define SEQ   2048
#define MROWS (BATCH*SEQ)   // 4096

typedef __attribute__((ext_vector_type(16))) __bf16        v16bf;
typedef __attribute__((ext_vector_type(8)))  float         v8f;
typedef __attribute__((ext_vector_type(4))) unsigned int  u32x4;
typedef __attribute__((ext_vector_type(8)))  int           i32x8v;
typedef __attribute__((ext_vector_type(4)))  int           i32x4v;

#if defined(__has_builtin)
#if __has_builtin(__builtin_amdgcn_tensor_load_to_lds)
#define HAVE_TDM 1
#endif
#endif
#ifndef HAVE_TDM
#define HAVE_TDM 0
#endif

struct Frag32B { u32x4 lo, hi; };

__device__ __forceinline__ unsigned short f2bf(float f) {
  unsigned int u = __builtin_bit_cast(unsigned int, f);
  u += 0x7FFFu + ((u >> 16) & 1u);          // round-to-nearest-even
  return (unsigned short)(u >> 16);
}

// Load a 16x32 (or 32x16) WMMA fragment slice for one lane:
// 8 contiguous bf16 at [k0] and 8 at [k0+16] (ISA 16-bit A/B VGPR layout).
__device__ __forceinline__ v16bf load_frag(const unsigned short* rowBase, int k0) {
  Frag32B t;
  t.lo = *(const u32x4*)(rowBase + k0);
  t.hi = *(const u32x4*)(rowBase + k0 + 16);
  return __builtin_bit_cast(v16bf, t);
}

__device__ __forceinline__ v8f wmma_bf16(v16bf a, v16bf b, v8f c) {
  return __builtin_amdgcn_wmma_f32_16x16x32_bf16(false, a, false, b, (short)0, c,
                                                 false, false);
}

#if HAVE_TDM
__device__ __forceinline__ unsigned rflu(unsigned v) {
  return (unsigned)__builtin_amdgcn_readfirstlane((int)v);
}

// TDM 2-D tile DMA: global (row-major, rowStride_u32 in 4-byte units) -> LDS
// (compact). Tile dims == tensor dims (no partial tiles at these sizes).
// D# packing per CDNA5 ISA 8.3/8.4: group0 = count|lds_addr|global_addr|type=2,
// group1 = data_size(4B) | tensor_dim0/1 | tile_dim0/1 | dim0_stride.
// 6-arg builtin form (clang-23 / therock headers): groups 0..3 + extra + cpol.
__device__ __forceinline__ void tdm_load_2d(unsigned ldsByteOff, const void* gptr,
                                            unsigned tileW_u32, unsigned tileH,
                                            unsigned rowStride_u32) {
  unsigned long long ga = (unsigned long long)gptr;
  u32x4 g0;
  g0.x = rflu(1u);                                     // count=1, user mode
  g0.y = rflu(ldsByteOff);                             // lds_addr (bytes)
  g0.z = rflu((unsigned)ga);                           // global_addr[31:0]
  g0.w = rflu((unsigned)((ga >> 32) & 0x1FFFFFFu) | 0x80000000u); // [56:32]|type=2
  i32x8v g1;
  g1[0] = (int)rflu(2u << 16);                         // data_size=4B, wg_mask=0
  g1[1] = (int)rflu((tileW_u32 & 0xFFFFu) << 16);      // tensor_dim0[15:0]
  g1[2] = (int)rflu(((tileW_u32 >> 16) & 0xFFFFu) |
                    ((tileH & 0xFFFFu) << 16));        // dim0 hi | dim1 lo
  g1[3] = (int)rflu(((tileH >> 16) & 0xFFFFu) |
                    ((tileW_u32 & 0xFFFFu) << 16));    // dim1 hi | tile_dim0
  g1[4] = (int)rflu(tileH & 0xFFFFu);                  // tile_dim1, tile_dim2=0
  g1[5] = (int)rflu(rowStride_u32);                    // dim0_stride[31:0]
  g1[6] = 0;                                           // stride hi | dim1_stride
  g1[7] = 0;
  i32x4v z4 = {0, 0, 0, 0};
  i32x8v z8 = {0, 0, 0, 0, 0, 0, 0, 0};
  __builtin_amdgcn_tensor_load_to_lds(g0, g1, z4, z4, z8, 0);
}
#endif

// ---------------------------------------------------------------- convert
__global__ void k_cvt_bf16(const float* __restrict__ in,
                           unsigned short* __restrict__ out, int n8) {
  int i = blockIdx.x * blockDim.x + threadIdx.x;
  if (i >= n8) return;
  const float4* p = (const float4*)in + (size_t)i * 2;
  float4 a = p[0], b = p[1];
  u32x4 v;
  v.x = (unsigned)f2bf(a.x) | ((unsigned)f2bf(a.y) << 16);
  v.y = (unsigned)f2bf(a.z) | ((unsigned)f2bf(a.w) << 16);
  v.z = (unsigned)f2bf(b.x) | ((unsigned)f2bf(b.y) << 16);
  v.w = (unsigned)f2bf(b.z) | ((unsigned)f2bf(b.w) << 16);
  *(u32x4*)(out + (size_t)i * 8) = v;
}

// ---------------------------------------------------------------- NT GEMM
// C[m,n] = sum_k A[m,k] * Bw[n,k]   (y = x @ W.T)
// mode 0/1: bf16 out scattered to [B,H,T,D]   (Q, K)
// mode 2  : bf16 out scattered to [B,H,D,T]   (V transposed)
// mode 3  : fp32 out row-major [M,N]          (final projection)
// Dynamic LDS (32 KB): A tiles at 0/8192, B tiles at 16384/24576 (128x32 bf16).
__global__ __launch_bounds__(256)
void k_gemm_nt(const unsigned short* __restrict__ A,
               const unsigned short* __restrict__ Bw,
               unsigned short* __restrict__ obf,
               float* __restrict__ ofp,
               int M, int N, int K, int mode) {
  extern __shared__ char smem[];

  const int tid   = threadIdx.x;
  const int lane  = tid & 31, w = tid >> 5;
  const int waveM = w >> 1, waveN = w & 1;        // 4 x 2 wave grid
  const int l15   = lane & 15, hl = lane >> 4;
  const int k0    = hl * 8;
  const int bm    = blockIdx.y * 128, bn = blockIdx.x * 128;

  v8f acc[2][4];
  #pragma unroll
  for (int mt = 0; mt < 2; ++mt)
    #pragma unroll
    for (int nt = 0; nt < 4; ++nt) acc[mt][nt] = (v8f)0.0f;

  const int NKB = K / 32;
#if HAVE_TDM
  const unsigned strideU = (unsigned)(K / 2);     // row stride in 4B units
  if (w == 0) {                                   // one wave drives the TDM
    tdm_load_2d(0u,     A  + (size_t)bm * K, 16u, 128u, strideU);
    tdm_load_2d(16384u, Bw + (size_t)bn * K, 16u, 128u, strideU);
  }
#endif

  for (int kb = 0; kb < NKB; ++kb) {
    const int buf = kb & 1;
    unsigned short* lA = (unsigned short*)(smem + (buf ? 8192 : 0));
    unsigned short* lB = (unsigned short*)(smem + (buf ? 24576 : 16384));
#if HAVE_TDM
    if (w == 0) {
      if (kb + 1 < NKB) {                         // prefetch next K-slab
        tdm_load_2d(buf ? 0u : 8192u,
                    A + (size_t)bm * K + (size_t)(kb + 1) * 32, 16u, 128u, strideU);
        tdm_load_2d(buf ? 16384u : 24576u,
                    Bw + (size_t)bn * K + (size_t)(kb + 1) * 32, 16u, 128u, strideU);
        __builtin_amdgcn_s_wait_tensorcnt(2);     // oldest 2 (current buf) done
      } else {
        __builtin_amdgcn_s_wait_tensorcnt(0);
      }
    }
    __syncthreads();
#else
    #pragma unroll
    for (int i = 0; i < 2; ++i) {
      int chunk = tid * 2 + i;                    // 0..511
      int row = chunk >> 2, off = (chunk & 3) * 8;
      *(u32x4*)&lA[row * 32 + off] =
          *(const u32x4*)&A[(size_t)(bm + row) * K + (size_t)kb * 32 + off];
      *(u32x4*)&lB[row * 32 + off] =
          *(const u32x4*)&Bw[(size_t)(bn + row) * K + (size_t)kb * 32 + off];
    }
    if (kb + 1 < NKB) {
      __builtin_prefetch(&A[(size_t)(bm + (tid >> 1)) * K + (size_t)kb * 32 + 32], 0, 1);
      __builtin_prefetch(&Bw[(size_t)(bn + (tid >> 1)) * K + (size_t)kb * 32 + 32], 0, 1);
    }
    __syncthreads();
#endif
    v16bf af[2], bfr[4];
    #pragma unroll
    for (int mt = 0; mt < 2; ++mt)
      af[mt] = load_frag(&lA[(waveM * 32 + mt * 16 + l15) * 32], k0);
    #pragma unroll
    for (int nt = 0; nt < 4; ++nt)
      bfr[nt] = load_frag(&lB[(waveN * 64 + nt * 16 + l15) * 32], k0);
    #pragma unroll
    for (int mt = 0; mt < 2; ++mt)
      #pragma unroll
      for (int nt = 0; nt < 4; ++nt)
        acc[mt][nt] = wmma_bf16(af[mt], bfr[nt], acc[mt][nt]);
    __syncthreads();
  }

  #pragma unroll
  for (int mt = 0; mt < 2; ++mt)
    #pragma unroll
    for (int nt = 0; nt < 4; ++nt)
      #pragma unroll
      for (int r = 0; r < 8; ++r) {
        int gm = bm + waveM * 32 + mt * 16 + r + hl * 8;   // C layout rows
        int gn = bn + waveN * 64 + nt * 16 + l15;          // C layout cols
        float v = acc[mt][nt][r];
        if (mode == 3) {
          ofp[(size_t)gm * N + gn] = v;
        } else {
          int b = gm / SEQ, t = gm % SEQ;
          int h = gn / HDIM, d = gn % HDIM;
          size_t idx = (mode == 2)
              ? ((size_t)(b * NHEAD + h) * HDIM + d) * SEQ + t   // V^T
              : ((size_t)(b * NHEAD + h) * SEQ + t) * HDIM + d;  // Q / K
          obf[idx] = f2bf(v);
        }
      }
}

// ---------------------------------------------------------------- attention
// Flash-style online softmax. grid = (SEQ/128, B*H), 256 threads (8 waves).
// Each wave: 16 query rows, full D=128 output, Q frags register-resident.
// Dynamic LDS (40 KB): K tiles 0/8192 (32x128), V^T tiles 16384/24576 (128x32),
// P scratch at 32768 (8 waves x 16x32 bf16).
__global__ __launch_bounds__(256)
void k_attn(const unsigned short* __restrict__ Q,    // [B*H, T, D]
            const unsigned short* __restrict__ Kk,   // [B*H, T, D]
            const unsigned short* __restrict__ Vt,   // [B*H, D, T]
            unsigned short* __restrict__ Y) {        // [B, T, C] bf16
  extern __shared__ char smem[];

  const int tid  = threadIdx.x;
  const int lane = tid & 31, w = tid >> 5;
  const int l15  = lane & 15, hl = lane >> 4;
  const int k0   = hl * 8;
  const int bh   = blockIdx.y;
  const int t0   = blockIdx.x * 128 + w * 16;
  const float scale = 0.08838834764831845f;          // 1/sqrt(128)

  unsigned short* lP = (unsigned short*)(smem + 32768);

  const unsigned short* Qb = Q  + (size_t)bh * SEQ * HDIM;
  const unsigned short* Kb = Kk + (size_t)bh * SEQ * HDIM;
  const unsigned short* Vb = Vt + (size_t)bh * HDIM * SEQ;

  v16bf qf[4];
  #pragma unroll
  for (int ks = 0; ks < 4; ++ks)
    qf[ks] = load_frag(Qb + (size_t)(t0 + l15) * HDIM + ks * 32, k0);

  v8f acc[8];
  #pragma unroll
  for (int dt = 0; dt < 8; ++dt) acc[dt] = (v8f)0.0f;
  float m_r[8], l_r[8];
  #pragma unroll
  for (int r = 0; r < 8; ++r) { m_r[r] = -3.0e38f; l_r[r] = 0.0f; }

  const int NSB = SEQ / 32;
#if HAVE_TDM
  if (w == 0) {
    tdm_load_2d(0u,     Kb, 64u, 32u, (unsigned)(HDIM / 2));  // 32 keys x 128d
    tdm_load_2d(16384u, Vb, 16u, 128u, (unsigned)(SEQ / 2));  // 128d x 32 keys
  }
#endif

  for (int sb = 0; sb < NSB; ++sb) {
    const int s0 = sb * 32;
    const int buf = sb & 1;
    unsigned short* lK = (unsigned short*)(smem + (buf ? 8192 : 0));
    unsigned short* lV = (unsigned short*)(smem + (buf ? 24576 : 16384));
#if HAVE_TDM
    if (w == 0) {
      if (sb + 1 < NSB) {
        tdm_load_2d(buf ? 0u : 8192u,
                    Kb + (size_t)(s0 + 32) * HDIM, 64u, 32u, (unsigned)(HDIM / 2));
        tdm_load_2d(buf ? 16384u : 24576u,
                    Vb + (size_t)(s0 + 32), 16u, 128u, (unsigned)(SEQ / 2));
        __builtin_amdgcn_s_wait_tensorcnt(2);
      } else {
        __builtin_amdgcn_s_wait_tensorcnt(0);
      }
    }
    __syncthreads();
#else
    #pragma unroll
    for (int i = 0; i < 2; ++i) {
      int chunk = tid * 2 + i;                       // 0..511
      { int row = chunk >> 4, off = (chunk & 15) * 8;
        *(u32x4*)&lK[row * 128 + off] =
            *(const u32x4*)&Kb[(size_t)(s0 + row) * HDIM + off]; }
      { int row = chunk >> 2, off = (chunk & 3) * 8;
        *(u32x4*)&lV[row * 32 + off] =
            *(const u32x4*)&Vb[(size_t)row * SEQ + s0 + off]; }
    }
    __syncthreads();
#endif

    // S = Q K^T for 32 keys (two 16-wide tiles), K-dim = 128 (4 WMMA steps)
    v8f sc0 = (v8f)0.0f, sc1 = (v8f)0.0f;
    #pragma unroll
    for (int ks = 0; ks < 4; ++ks) {
      v16bf b0 = load_frag(&lK[(l15) * 128 + ks * 32], k0);
      v16bf b1 = load_frag(&lK[(16 + l15) * 128 + ks * 32], k0);
      sc0 = wmma_bf16(qf[ks], b0, sc0);
      sc1 = wmma_bf16(qf[ks], b1, sc1);
    }

    // online softmax: row = r + 8*hl; cols across the 16 lanes of each half
    float corr[8];
    #pragma unroll
    for (int r = 0; r < 8; ++r) {
      float s0v = sc0[r] * scale, s1v = sc1[r] * scale;
      float mx = fmaxf(s0v, s1v);
      #pragma unroll
      for (int off = 8; off; off >>= 1) mx = fmaxf(mx, __shfl_xor(mx, off, 16));
      float mnew = fmaxf(m_r[r], mx);
      float c  = __expf(m_r[r] - mnew);
      float p0 = __expf(s0v - mnew);
      float p1 = __expf(s1v - mnew);
      float rs = p0 + p1;
      #pragma unroll
      for (int off = 8; off; off >>= 1) rs += __shfl_xor(rs, off, 16);
      l_r[r] = l_r[r] * c + rs;
      m_r[r] = mnew;
      corr[r] = c;
      int prow = r + hl * 8;
      lP[(w * 16 + prow) * 32 + l15]      = f2bf(p0); // C-layout -> A-layout LDS
      lP[(w * 16 + prow) * 32 + l15 + 16] = f2bf(p1);
    }
    #pragma unroll
    for (int dt = 0; dt < 8; ++dt)
      #pragma unroll
      for (int r = 0; r < 8; ++r) acc[dt][r] *= corr[r];

    // wave-local LDS round trip: wait our DS stores, no workgroup barrier
    asm volatile("s_wait_dscnt 0" ::: "memory");

    v16bf pf = load_frag(&lP[(w * 16 + l15) * 32], k0);
    #pragma unroll
    for (int dt = 0; dt < 8; ++dt) {
      v16bf vf = load_frag(&lV[(dt * 16 + l15) * 32], k0);
      acc[dt] = wmma_bf16(pf, vf, acc[dt]);
    }
    __syncthreads();
  }

  const int b = bh >> 4, h = bh & 15;
  #pragma unroll
  for (int dt = 0; dt < 8; ++dt)
    #pragma unroll
    for (int r = 0; r < 8; ++r) {
      int t = t0 + r + hl * 8;
      int d = dt * 16 + l15;
      float o = acc[dt][r] / l_r[r];
      Y[((size_t)(b * SEQ + t)) * DIMC + h * HDIM + d] = f2bf(o);
    }
}

// ---------------------------------------------------------------- launcher
extern "C" void kernel_launch(void* const* d_in, const int* in_sizes, int n_in,
                              void* d_out, int out_size, void* d_ws, size_t ws_size,
                              hipStream_t stream) {
  const float* x  = (const float*)d_in[0];
  const float* wq = (const float*)d_in[1];
  const float* wk = (const float*)d_in[2];
  const float* wv = (const float*)d_in[3];
  const float* wo = (const float*)d_in[4];
  float* out = (float*)d_out;

  char* ws = (char*)d_ws;
  const size_t MB = 1024 * 1024;
  unsigned short* xb  = (unsigned short*)(ws + 0 * MB);    // 16 MB
  unsigned short* wqb = (unsigned short*)(ws + 16 * MB);   //  8 MB
  unsigned short* wkb = (unsigned short*)(ws + 24 * MB);
  unsigned short* wvb = (unsigned short*)(ws + 32 * MB);
  unsigned short* wob = (unsigned short*)(ws + 40 * MB);
  unsigned short* Qb  = (unsigned short*)(ws + 48 * MB);   // 16 MB [B,H,T,D]
  unsigned short* Kb  = (unsigned short*)(ws + 64 * MB);   // 16 MB [B,H,T,D]
  unsigned short* Vtb = (unsigned short*)(ws + 80 * MB);   // 16 MB [B,H,D,T]
  unsigned short* Yb  = (unsigned short*)(ws + 96 * MB);   // 16 MB [B,T,C]

  const int nx8 = BATCH * SEQ * DIMC / 8;
  const int nw8 = DIMC * DIMC / 8;
  k_cvt_bf16<<<(nx8 + 255) / 256, 256, 0, stream>>>(x,  xb,  nx8);
  k_cvt_bf16<<<(nw8 + 255) / 256, 256, 0, stream>>>(wq, wqb, nw8);
  k_cvt_bf16<<<(nw8 + 255) / 256, 256, 0, stream>>>(wk, wkb, nw8);
  k_cvt_bf16<<<(nw8 + 255) / 256, 256, 0, stream>>>(wv, wvb, nw8);
  k_cvt_bf16<<<(nw8 + 255) / 256, 256, 0, stream>>>(wo, wob, nw8);

  dim3 gg(DIMC / 128, MROWS / 128);   // (16, 32)
  k_gemm_nt<<<gg, 256, 32768, stream>>>(xb, wqb, Qb,  nullptr, MROWS, DIMC, DIMC, 0);
  k_gemm_nt<<<gg, 256, 32768, stream>>>(xb, wkb, Kb,  nullptr, MROWS, DIMC, DIMC, 1);
  k_gemm_nt<<<gg, 256, 32768, stream>>>(xb, wvb, Vtb, nullptr, MROWS, DIMC, DIMC, 2);

  dim3 ga(SEQ / 128, BATCH * NHEAD);  // (16, 32)
  k_attn<<<ga, 256, 40960, stream>>>(Qb, Kb, Vtb, Yb);

  k_gemm_nt<<<gg, 256, 32768, stream>>>(Yb, wob, nullptr, out, MROWS, DIMC, DIMC, 3);
}
